// RNN_33337536152297
// MI455X (gfx1250) — matrix operand order repaired and checked by reading, not compiled, not run on hardware
//
#include <hip/hip_runtime.h>
#include <hip/hip_bf16.h>

// ---------------------------------------------------------------------------
// LSTM (L=2, B=64, T=256, H=1024, V=32000) with bf16 WMMA on gfx1250.
// ---------------------------------------------------------------------------

#define Hh   1024
#define Bb   64
#define Tt   256
#define Vv   32000
#define G4   4096   // 4*H

typedef __attribute__((ext_vector_type(16))) __bf16 v16bf;
typedef __attribute__((ext_vector_type(8)))  float  v8f;

using bf16_t = __hip_bfloat16;

static __device__ inline v8f v8f_zero() {
    v8f z = {0.f, 0.f, 0.f, 0.f, 0.f, 0.f, 0.f, 0.f};
    return z;
}

// Assemble a 16-element bf16 fragment from two 16-byte chunks.
static __device__ inline v16bf frag16(const void* p0, const void* p1) {
    v16bf r;
    __builtin_memcpy(&r, p0, 16);
    __builtin_memcpy(reinterpret_cast<char*>(&r) + 16, p1, 16);
    return r;
}

static __device__ inline v8f wmma_bf16(v16bf a, v16bf b, v8f c) {
    return __builtin_amdgcn_wmma_f32_16x16x32_bf16(
        /*neg_a=*/false, a, /*neg_b=*/false, b,
        /*c_mod=*/(short)0, c, /*reuse_a=*/false, /*reuse_b=*/false);
}

// Async global->LDS 16-byte copy (CDNA5). lds_off = workgroup LDS byte offset
// (low 32 bits of the generic pointer to a __shared__ object). Tracked by
// ASYNCcnt; pair with s_wait_asynccnt + barrier before consuming.
static __device__ inline void async_copy16(unsigned int lds_off, const void* gptr) {
    asm volatile("global_load_async_to_lds_b128 %0, %1, off"
                 :: "v"(lds_off), "v"(gptr)
                 : "memory");
}
static __device__ inline void wait_async_le1() {
    asm volatile("s_wait_asynccnt 0x1" ::: "memory");
}
static __device__ inline void wait_async_le0() {
    asm volatile("s_wait_asynccnt 0x0" ::: "memory");
}

static __device__ inline float sigmoidf_(float x) {
    return 1.0f / (1.0f + __expf(-x));
}

// ---------------------------------------------------------------------------
// Small utility kernels
// ---------------------------------------------------------------------------
__global__ void cast_f2b_kernel(const float* __restrict__ s,
                                bf16_t* __restrict__ d, long n) {
    long i = (long)blockIdx.x * 256 + threadIdx.x;
    if (i < n) d[i] = __float2bfloat16(s[i]);
}

__global__ void bias_sum_kernel(const float* __restrict__ a,
                                const float* __restrict__ b,
                                float* __restrict__ d, int n) {
    int i = blockIdx.x * 256 + threadIdx.x;
    if (i < n) d[i] = a[i] + b[i];
}

__global__ void zero_f32_kernel(float* __restrict__ p, int n) {
    int i = blockIdx.x * 256 + threadIdx.x;
    if (i < n) p[i] = 0.0f;
}

__global__ void zero_b16_kernel(bf16_t* __restrict__ p, int n) {
    int i = blockIdx.x * 256 + threadIdx.x;
    if (i < n) p[i] = __float2bfloat16(0.0f);
}

// x_bf[bt][h] = bf16(embed[inputs[bt]][h])
__global__ void embed_cast_kernel(const int* __restrict__ inputs,
                                  const float* __restrict__ emb,
                                  bf16_t* __restrict__ xb, long n) {
    long i = (long)blockIdx.x * 256 + threadIdx.x;
    if (i < n) {
        long bt = i >> 10;          // / H
        int  h  = (int)(i & 1023);  // % H
        int tok = inputs[bt];
        xb[i] = __float2bfloat16(emb[(size_t)tok * Hh + h]);
    }
}

// ---------------------------------------------------------------------------
// Generic bf16 WMMA GEMM:  C[M,N] = A[M,K] * B[N,K]^T + bias[N]   (fp32 out)
// Block tile 64(M) x 256(N), 256 threads = 8 waves, each wave 4Mx2N 16x16 tiles.
// A tile double-buffered in LDS via async global->LDS copies; the steady-state
// loop processes tile pairs so buffer roles are compile-time constant.
// Requires M % 64 == 0, N % 256 == 0; K is a template param, K % 64 == 0.
// ---------------------------------------------------------------------------
template <int K>
__global__ __launch_bounds__(256)
void gemm_bf16_wmma(const bf16_t* __restrict__ A, long lda,
                    const bf16_t* __restrict__ B,
                    const float* __restrict__ bias,
                    float* __restrict__ C, long ldc) {
    static_assert(K % 64 == 0 && K >= 128, "K must be a multiple of 64");
    constexpr int NITER = K / 32;

    __shared__ bf16_t Abuf[2][64 * 32];

    const int tid  = threadIdx.x;
    const int wave = tid >> 5;
    const int lane = tid & 31;
    const int lh   = lane >> 4;   // 0/1 (lane half)
    const int lm   = lane & 15;

    const long m0 = (long)blockIdx.y * 64;
    const long n0 = (long)blockIdx.x * 256;

    v8f acc[4][2];
#pragma unroll
    for (int i = 0; i < 4; ++i)
#pragma unroll
        for (int j = 0; j < 2; ++j) acc[i][j] = v8f_zero();

    // per-wave B row base pointers (N,K row-major)
    const bf16_t* Bbase[2];
#pragma unroll
    for (int nt = 0; nt < 2; ++nt) {
        long n = n0 + (long)(wave * 2 + nt) * 16 + lm;
        Bbase[nt] = B + n * (long)K + lh * 8;
    }

    const int srow = tid >> 2;          // 0..63
    const int sseg = (tid & 3) * 8;     // 0,8,16,24
    const bf16_t* Asrc = A + (m0 + srow) * lda + sseg;
    const unsigned int lds0 = (unsigned int)(uintptr_t)&Abuf[0][srow * 32 + sseg];
    const unsigned int lds1 = (unsigned int)(uintptr_t)&Abuf[1][srow * 32 + sseg];

    const bf16_t* a0 = &Abuf[0][(0 * 16 + lm) * 32 + lh * 8];  // per-lane A frag base (buf0)
    const bf16_t* a1 = &Abuf[1][(0 * 16 + lm) * 32 + lh * 8];

    auto compute = [&](const bf16_t* abase, int k0) {
        v16bf a[4];
#pragma unroll
        for (int mt = 0; mt < 4; ++mt) {
            const bf16_t* ar = abase + mt * (16 * 32);
            a[mt] = frag16(ar, ar + 16);
        }
#pragma unroll
        for (int nt = 0; nt < 2; ++nt) {
            v16bf b = frag16(Bbase[nt] + k0, Bbase[nt] + k0 + 16);
#pragma unroll
            for (int mt = 0; mt < 4; ++mt)
                acc[mt][nt] = wmma_bf16(a[mt], b, acc[mt][nt]);
        }
    };

    async_copy16(lds0, Asrc);           // prefetch tile 0 into buf0

    int i = 0;
    for (; i < NITER - 2; i += 2) {
        __syncthreads();                              // buf1 free
        async_copy16(lds1, Asrc + (i + 1) * 32);
        wait_async_le1();                             // own tile i done
        __syncthreads();                              // all tile i done
        compute(a0, i * 32);
        __syncthreads();                              // buf0 free
        async_copy16(lds0, Asrc + (i + 2) * 32);
        wait_async_le1();                             // own tile i+1 done
        __syncthreads();                              // all tile i+1 done
        compute(a1, (i + 1) * 32);
    }
    // tail pair: i == NITER-2
    __syncthreads();
    async_copy16(lds1, Asrc + (i + 1) * 32);
    wait_async_le1();
    __syncthreads();
    compute(a0, i * 32);
    __syncthreads();
    wait_async_le0();
    __syncthreads();
    compute(a1, (i + 1) * 32);

#pragma unroll
    for (int nt = 0; nt < 2; ++nt) {
        long n = n0 + (long)(wave * 2 + nt) * 16 + lm;
        float bv = bias ? bias[n] : 0.0f;
#pragma unroll
        for (int mt = 0; mt < 4; ++mt)
#pragma unroll
            for (int r = 0; r < 8; ++r) {
                long m = m0 + mt * 16 + r + 8 * lh;
                C[m * ldc + n] = acc[mt][nt][r] + bv;
            }
    }
}

// ---------------------------------------------------------------------------
// One LSTM timestep: gates = xg[:,t,:] + h_prev @ Whh^T, then cell update.
// Grid: 64 blocks; block owns h-slice j0..j0+15 across all 4 gates
// (64 batch x 64 gate-cols output, K=1024). 256 threads = 8 waves,
// wave w: M-tile w/2, N-tiles {2*(w&1), 2*(w&1)+1}.
// A fragments load straight from global (h_prev is tiny, cache resident) ->
// zero barriers in the K loop, which is the serial critical path. xg and c
// loads are hoisted above the GEMM so they are in flight during it.
// ---------------------------------------------------------------------------
__global__ __launch_bounds__(256)
void lstm_step_kernel(const float* __restrict__ xg,       // (B*T, 4H)
                      const bf16_t* __restrict__ Whh,     // (4H, H) bf16
                      const bf16_t* __restrict__ h_prev,  // (B, H) bf16
                      bf16_t* __restrict__ h_next,        // (B, H) bf16
                      float* __restrict__ c,              // (B, H) fp32
                      bf16_t* __restrict__ ys,            // (B, T, H) bf16
                      float* __restrict__ hn_out,         // (B, H) at t==T-1
                      float* __restrict__ cn_out,         // (B, H) at t==T-1
                      float* __restrict__ last_out,       // (B, H) or null
                      int t) {
    __shared__ float Gts[64 * 64];   // [batch][g*16 + jj]

    const int tid  = threadIdx.x;
    const int wave = tid >> 5;
    const int lane = tid & 31;
    const int lh   = lane >> 4;
    const int lm   = lane & 15;
    const int j0   = blockIdx.x * 16;

    const int mt  = wave >> 1;        // 0..3 (M tile)
    const int ntb = (wave & 1) * 2;   // 0 or 2 (first N tile)

    // Preload input-gate contributions (in flight during the GEMM)
    float xgv[2][8];
#pragma unroll
    for (int q = 0; q < 2; ++q) {
        int colc = (ntb + q) * 16 + lm;
        int g = colc >> 4, jj = colc & 15;
        int gcol = g * Hh + j0 + jj;
#pragma unroll
        for (int r = 0; r < 8; ++r) {
            int b = mt * 16 + r + 8 * lh;
            xgv[q][r] = xg[((size_t)b * Tt + t) * G4 + gcol];
        }
    }
    // Preload this thread's cell-state values (independent of the GEMM)
    float cp[4];
#pragma unroll
    for (int it = 0; it < 4; ++it) {
        int id = tid + 256 * it;
        cp[it] = c[(id >> 4) * Hh + j0 + (id & 15)];
    }

    v8f acc[2];
    acc[0] = v8f_zero();
    acc[1] = v8f_zero();

    // A fragment base: row (mt*16+lm) of h_prev, chunks at k {lh*8, 16+lh*8}
    const bf16_t* Arow = h_prev + (mt * 16 + lm) * Hh + lh * 8;
    const bf16_t* Bbase[2];
#pragma unroll
    for (int q = 0; q < 2; ++q) {
        int colc = (ntb + q) * 16 + lm;
        int g = colc >> 4, jj = colc & 15;
        Bbase[q] = Whh + (size_t)(g * Hh + j0 + jj) * Hh + lh * 8;
    }

#pragma unroll 4
    for (int k0 = 0; k0 < Hh; k0 += 32) {
        v16bf a = frag16(Arow + k0, Arow + k0 + 16);
#pragma unroll
        for (int q = 0; q < 2; ++q) {
            v16bf b = frag16(Bbase[q] + k0, Bbase[q] + k0 + 16);
            acc[q] = wmma_bf16(a, b, acc[q]);
        }
    }

    // accumulate input-gate terms and park in LDS for the cell phase
#pragma unroll
    for (int q = 0; q < 2; ++q) {
        int colc = (ntb + q) * 16 + lm;
#pragma unroll
        for (int r = 0; r < 8; ++r) {
            int b = mt * 16 + r + 8 * lh;
            Gts[b * 64 + colc] = acc[q][r] + xgv[q][r];
        }
    }
    __syncthreads();

    // cell update: 64 batch x 16 units = 1024 items, 4 per thread
#pragma unroll
    for (int it = 0; it < 4; ++it) {
        int id = tid + 256 * it;
        int b  = id >> 4;
        int jj = id & 15;
        float ig = sigmoidf_(Gts[b * 64 + 0 * 16 + jj]);
        float fg = sigmoidf_(Gts[b * 64 + 1 * 16 + jj]);
        float gg = tanhf(Gts[b * 64 + 2 * 16 + jj]);
        float og = sigmoidf_(Gts[b * 64 + 3 * 16 + jj]);
        int hidx = b * Hh + j0 + jj;
        float cn = fg * cp[it] + ig * gg;
        float hv = og * tanhf(cn);
        c[hidx] = cn;
        bf16_t hb = __float2bfloat16(hv);
        h_next[hidx] = hb;
        ys[((size_t)b * Tt + t) * Hh + j0 + jj] = hb;
        if (t == Tt - 1) {
            hn_out[hidx] = hv;
            cn_out[hidx] = cn;
            if (last_out) last_out[hidx] = hv;
        }
    }
}

// ---------------------------------------------------------------------------
// Host orchestration
// ---------------------------------------------------------------------------
extern "C" void kernel_launch(void* const* d_in, const int* in_sizes, int n_in,
                              void* d_out, int out_size, void* d_ws, size_t ws_size,
                              hipStream_t stream) {
    (void)in_sizes; (void)n_in; (void)out_size; (void)ws_size;

    const int*   inputs = (const int*)d_in[0];    // (B, T)
    const float* embed  = (const float*)d_in[1];  // (V, H)
    const float* W_ih   = (const float*)d_in[2];  // (L, 4H, H)
    const float* W_hh   = (const float*)d_in[3];  // (L, 4H, H)
    const float* b_ih   = (const float*)d_in[4];  // (L, 4H)
    const float* b_hh   = (const float*)d_in[5];  // (L, 4H)
    const float* W_proj = (const float*)d_in[6];  // (V, H)
    const float* b_proj = (const float*)d_in[7];  // (V)

    float* out        = (float*)d_out;
    float* out_last   = out;                              // 64*1024
    float* out_logits = out + (size_t)Bb * Hh;            // 64*32000
    float* out_hn     = out_logits + (size_t)Bb * Vv;     // 2*64*1024
    float* out_cn     = out_hn + (size_t)2 * Bb * Hh;     // 2*64*1024

    char* wsp = (char*)d_ws;
    auto alloc = [&](size_t bytes) -> char* {
        char* p = wsp;
        wsp += (bytes + 255) & ~(size_t)255;
        return p;
    };
    bf16_t* x_bf   = (bf16_t*)alloc((size_t)Bb * Tt * Hh * 2);       //  33.5 MB
    float*  xg     = (float*) alloc((size_t)Bb * Tt * G4 * 4);       // 268   MB
    bf16_t* wih_bf = (bf16_t*)alloc((size_t)2 * G4 * Hh * 2);        //  16.8 MB
    bf16_t* whh_bf = (bf16_t*)alloc((size_t)2 * G4 * Hh * 2);        //  16.8 MB
    bf16_t* wpr_bf = (bf16_t*)alloc((size_t)Vv * Hh * 2);            //  65.5 MB
    float*  bsum   = (float*) alloc((size_t)2 * G4 * 4);
    bf16_t* hping  = (bf16_t*)alloc((size_t)Bb * Hh * 2);
    bf16_t* hpong  = (bf16_t*)alloc((size_t)Bb * Hh * 2);
    float*  cbuf   = (float*) alloc((size_t)Bb * Hh * 4);

    auto blocks = [](long n) { return (unsigned)((n + 255) / 256); };

    // 1) casts to bf16
    {
        long n = (long)2 * G4 * Hh;
        cast_f2b_kernel<<<blocks(n), 256, 0, stream>>>(W_ih, wih_bf, n);
        cast_f2b_kernel<<<blocks(n), 256, 0, stream>>>(W_hh, whh_bf, n);
        long np = (long)Vv * Hh;
        cast_f2b_kernel<<<blocks(np), 256, 0, stream>>>(W_proj, wpr_bf, np);
        bias_sum_kernel<<<blocks(2 * G4), 256, 0, stream>>>(b_ih, b_hh, bsum, 2 * G4);
        long ne = (long)Bb * Tt * Hh;
        embed_cast_kernel<<<blocks(ne), 256, 0, stream>>>(inputs, embed, x_bf, ne);
    }

    // 2) layers
    for (int l = 0; l < 2; ++l) {
        zero_f32_kernel<<<blocks(Bb * Hh), 256, 0, stream>>>(cbuf, Bb * Hh);
        zero_b16_kernel<<<blocks(Bb * Hh), 256, 0, stream>>>(hping, Bb * Hh);

        // xg = x @ W_ih[l]^T + (b_ih[l]+b_hh[l]) :  M=16384, N=4096, K=1024
        gemm_bf16_wmma<Hh><<<dim3(G4 / 256, (Bb * Tt) / 64), 256, 0, stream>>>(
            x_bf, Hh,
            wih_bf + (size_t)l * G4 * Hh,
            bsum + (size_t)l * G4,
            xg, G4);

        bf16_t* hp = hping;
        bf16_t* hn = hpong;
        for (int t = 0; t < Tt; ++t) {
            lstm_step_kernel<<<dim3(Hh / 16), 256, 0, stream>>>(
                xg, whh_bf + (size_t)l * G4 * Hh, hp, hn, cbuf, x_bf,
                out_hn + (size_t)l * Bb * Hh,
                out_cn + (size_t)l * Bb * Hh,
                (l == 1) ? out_last : nullptr, t);
            bf16_t* tmp = hp; hp = hn; hn = tmp;
        }
    }

    // 3) logits = last_output @ W_proj^T + b_proj : M=64, N=32000, K=1024
    gemm_bf16_wmma<Hh><<<dim3(Vv / 256, Bb / 64), 256, 0, stream>>>(
        x_bf + (size_t)(Tt - 1) * Hh, (long)Tt * Hh,
        wpr_bf,
        b_proj,
        out_logits, Vv);
}
